// AttnDecoderRNN_678604833289
// MI455X (gfx1250) — matrix operand order repaired and checked by reading, not compiled, not run on hardware
//
#include <hip/hip_runtime.h>
#include <hip/hip_bf16.h>
#include <math.h>

#define Hd 1024
#define Vd 50000
#define Bd 64
#define Ld 256
#define EPSF 1e-7f

typedef __attribute__((ext_vector_type(16))) _Float16 v16h;
typedef __attribute__((ext_vector_type(8)))  _Float16 v8h;
typedef __attribute__((ext_vector_type(8)))  float    v8f;
typedef __attribute__((ext_vector_type(4)))  float    v4f;

// ---------------- embedding gather + relu ----------------
__global__ void k_embed(const int* __restrict__ ids, const float* __restrict__ emb,
                        float* __restrict__ x) {
    int b = blockIdx.x;
    const float* src = emb + (size_t)ids[b] * Hd;
    float* dst = x + (size_t)b * Hd;
    for (int h = threadIdx.x; h < Hd; h += blockDim.x)
        dst[h] = fmaxf(src[h], 0.0f);
}

// ---------------- WMMA GEMM: out[m,n] = sum_k A[m,k]*W[n,k] + bias[n] ----------------
// A: (M x K) f32 row-major (row stride K); W: (N x K) f32 row-major.
// grid.y tiles M by 16; grid.x tiles N by 128 (8 waves x 16 cols). block = 256 thr.
// dyn LDS = 16*(K+8) halfs.
__global__ void k_wmma_gemm_bt(const float* __restrict__ A, const float* __restrict__ W,
                               const float* __restrict__ bias, float* __restrict__ out,
                               int N, int K) {
    extern __shared__ __align__(16) _Float16 Alds[];
    const int ldA = K + 8;
    const int m0  = blockIdx.y * 16;

    // stage 16 x K A-tile into LDS as f16 (coalesced, converted)
    for (int r = 0; r < 16; ++r) {
        const float* arow = A + (size_t)(m0 + r) * K;
        for (int c = threadIdx.x; c < K; c += blockDim.x)
            Alds[r * ldA + c] = (_Float16)arow[c];
    }
    __syncthreads();

    const int lane = threadIdx.x & 31;
    const int wave = threadIdx.x >> 5;
    const int n0   = blockIdx.x * 128 + wave * 16;
    if (n0 >= N) return;                       // wave-uniform: EXEC stays all-1s

    const int row    = lane & 15;              // A: M row / B: N col / C: N col
    const int hiHalf = lane >> 4;              // 0: lanes 0-15, 1: lanes 16-31
    const int aoff   = hiHalf * 16;            // A frag K base (contiguous, matches B)
    const int boff   = hiHalf * 16;            // B frag K base (contiguous, matches A)
    const int n      = n0 + row;
    const float* wrow = W + (size_t)n * K;

    v8f acc = {};
#pragma unroll 4
    for (int kk = 0; kk < K; kk += 32) {
        // A fragment: halfs 0..15 = K[kk+aoff .. +15]
        const _Float16* ap = &Alds[row * ldA + kk + aoff];
        v8h lo = *(const v8h*)(ap);
        v8h hi = *(const v8h*)(ap + 8);
        v16h afrag = __builtin_shufflevector(lo, hi,
            0,1,2,3,4,5,6,7,8,9,10,11,12,13,14,15);

        // B fragment: 16 consecutive K values of w_out row n (streamed, non-temporal)
        v16h bfrag;
        const v4f* wp = (const v4f*)(wrow + kk + boff);
#pragma unroll
        for (int q = 0; q < 4; ++q) {
            v4f w4 = __builtin_nontemporal_load(wp + q);
#pragma unroll
            for (int j = 0; j < 4; ++j)
                bfrag[4 * q + j] = (_Float16)w4[j];
        }

        acc = __builtin_amdgcn_wmma_f32_16x16x32_f16(
            false, afrag, false, bfrag, (short)0, acc, false, false);
    }

    // C/D layout: lanes 0-15 -> M = m0 + i, lanes 16-31 -> M = m0 + 8 + i; N = n
    const int mbase = m0 + hiHalf * 8;
    const float bv = bias[n];
#pragma unroll
    for (int i = 0; i < 8; ++i)
        out[(size_t)(mbase + i) * N + n] = acc[i] + bv;
}

// ---------------- GRU gates ----------------
__global__ void k_gates(const float* __restrict__ gi, const float* __restrict__ gh,
                        const float* __restrict__ hid, float* __restrict__ cat,
                        float* __restrict__ hnew_out) {
    int idx = blockIdx.x * blockDim.x + threadIdx.x;
    if (idx >= Bd * Hd) return;
    int b = idx >> 10, j = idx & (Hd - 1);
    const float* gib = gi + (size_t)b * 3 * Hd;
    const float* ghb = gh + (size_t)b * 3 * Hd;
    float r = 1.0f / (1.0f + expf(-(gib[j] + ghb[j])));
    float z = 1.0f / (1.0f + expf(-(gib[Hd + j] + ghb[Hd + j])));
    float nn = tanhf(gib[2 * Hd + j] + r * ghb[2 * Hd + j]);
    float hv = hid[idx];
    float hn = (1.0f - z) * nn + z * hv;
    cat[(size_t)b * 2 * Hd + j] = hn;
    hnew_out[idx] = hn;
}

// ---------------- ||h_new[b]|| ----------------
__global__ void k_hnorm(const float* __restrict__ cat, float* __restrict__ hnorm) {
    __shared__ float sb[256];
    int b = blockIdx.x, t = threadIdx.x;
    const float* hb = cat + (size_t)b * 2 * Hd;
    float s = 0.0f;
    for (int j = t; j < Hd; j += 256) { float v = hb[j]; s += v * v; }
    sb[t] = s; __syncthreads();
    for (int st = 128; st > 0; st >>= 1) {
        if (t < st) sb[t] += sb[t + st];
        __syncthreads();
    }
    if (t == 0) hnorm[b] = fmaxf(sqrtf(sb[0]), EPSF);
}

// ---------------- cosine sim[l,b] ----------------
__global__ void k_sim(const float* __restrict__ cat, const float* __restrict__ enc,
                      const float* __restrict__ hnorm, float* __restrict__ sim) {
    __shared__ float sd[256], se[256];
    int l = blockIdx.x, b = blockIdx.y, t = threadIdx.x;
    const float* hb = cat + (size_t)b * 2 * Hd;
    const float* eb = enc + ((size_t)l * Bd + b) * Hd;
    float d = 0.0f, e = 0.0f;
    for (int j = t; j < Hd; j += 256) { float ev = eb[j]; d += hb[j] * ev; e += ev * ev; }
    sd[t] = d; se[t] = e; __syncthreads();
    for (int st = 128; st > 0; st >>= 1) {
        if (t < st) { sd[t] += sd[t + st]; se[t] += se[t + st]; }
        __syncthreads();
    }
    if (t == 0)
        sim[(size_t)l * Bd + b] = sd[0] / (hnorm[b] * fmaxf(sqrtf(se[0]), EPSF));
}

// ---------------- softmax over L + context; writes attn_weights & ctx half of cat ----------------
__global__ void k_softmax_ctx(const float* __restrict__ sim, const float* __restrict__ enc,
                              float* __restrict__ attn_out, float* __restrict__ cat) {
    __shared__ float sb[256];
    __shared__ float at[Ld];
    int b = blockIdx.x, t = threadIdx.x;      // t == l
    float s = sim[(size_t)t * Bd + b];
    sb[t] = s; __syncthreads();
    for (int st = 128; st > 0; st >>= 1) {
        if (t < st) sb[t] = fmaxf(sb[t], sb[t + st]);
        __syncthreads();
    }
    float mx = sb[0]; __syncthreads();
    float ex = expf(s - mx);
    sb[t] = ex; __syncthreads();
    for (int st = 128; st > 0; st >>= 1) {
        if (t < st) sb[t] += sb[t + st];
        __syncthreads();
    }
    float a = ex / sb[0];
    at[t] = a;
    attn_out[(size_t)b * Ld + t] = a;         // attn_weights is (B, L)
    __syncthreads();
    // ctx[b,h] = sum_l at[l] * enc[l,b,h]
    for (int h0 = 0; h0 < Hd; h0 += 256) {
        float accv = 0.0f;
        for (int l = 0; l < Ld; ++l)
            accv += at[l] * enc[((size_t)l * Bd + b) * Hd + h0 + t];
        cat[(size_t)b * 2 * Hd + Hd + h0 + t] = accv;
    }
}

// ---------------- in-place log-softmax over V ----------------
__global__ void k_logsoftmax(float* __restrict__ logits) {
    __shared__ float sb[256];
    int b = blockIdx.x, t = threadIdx.x;
    float* row = logits + (size_t)b * Vd;
    float m = -INFINITY;
    for (int j = t; j < Vd; j += 256) m = fmaxf(m, row[j]);
    sb[t] = m; __syncthreads();
    for (int st = 128; st > 0; st >>= 1) {
        if (t < st) sb[t] = fmaxf(sb[t], sb[t + st]);
        __syncthreads();
    }
    float mx = sb[0]; __syncthreads();
    float s = 0.0f;
    for (int j = t; j < Vd; j += 256) s += expf(row[j] - mx);
    sb[t] = s; __syncthreads();
    for (int st = 128; st > 0; st >>= 1) {
        if (t < st) sb[t] += sb[t + st];
        __syncthreads();
    }
    float lse = mx + logf(sb[0]);
    for (int j = t; j < Vd; j += 256) row[j] = row[j] - lse;
}

extern "C" void kernel_launch(void* const* d_in, const int* in_sizes, int n_in,
                              void* d_out, int out_size, void* d_ws, size_t ws_size,
                              hipStream_t stream) {
    (void)in_sizes; (void)n_in; (void)out_size; (void)ws_size;

    const int*   ids   = (const int*)  d_in[0];
    const float* hid   = (const float*)d_in[1];   // (1,B,H) contiguous
    const float* enc   = (const float*)d_in[2];   // (L,B,H)
    const float* emb   = (const float*)d_in[3];   // (V,H)
    const float* w_ih  = (const float*)d_in[4];   // (3H,H)
    const float* w_hh  = (const float*)d_in[5];   // (3H,H)
    const float* b_ih  = (const float*)d_in[6];
    const float* b_hh  = (const float*)d_in[7];
    const float* w_out = (const float*)d_in[8];   // (V,2H)
    const float* b_out = (const float*)d_in[9];

    float* lp   = (float*)d_out;                       // log_probs (B,V)
    float* hnew = lp + (size_t)Bd * Vd;                // h_new (B,H)
    float* attn = hnew + (size_t)Bd * Hd;              // attn_weights (B,L)

    float* ws    = (float*)d_ws;
    float* x     = ws;  ws += Bd * Hd;                 // relu(emb[ids])
    float* gi    = ws;  ws += Bd * 3 * Hd;
    float* gh    = ws;  ws += Bd * 3 * Hd;
    float* cat   = ws;  ws += Bd * 2 * Hd;             // [h_new | ctx]
    float* sim   = ws;  ws += Ld * Bd;
    float* hnorm = ws;  ws += Bd;

    k_embed<<<Bd, 256, 0, stream>>>(ids, emb, x);

    size_t shGRU = (size_t)16 * (Hd + 8) * sizeof(_Float16);
    k_wmma_gemm_bt<<<dim3(3 * Hd / 128, Bd / 16), 256, shGRU, stream>>>(
        x,   w_ih, b_ih, gi, 3 * Hd, Hd);
    k_wmma_gemm_bt<<<dim3(3 * Hd / 128, Bd / 16), 256, shGRU, stream>>>(
        hid, w_hh, b_hh, gh, 3 * Hd, Hd);

    k_gates<<<(Bd * Hd) / 256, 256, 0, stream>>>(gi, gh, hid, cat, hnew);
    k_hnorm<<<Bd, 256, 0, stream>>>(cat, hnorm);
    k_sim<<<dim3(Ld, Bd), 256, 0, stream>>>(cat, enc, hnorm, sim);
    k_softmax_ctx<<<Bd, 256, 0, stream>>>(sim, enc, attn, cat);

    size_t shOut = (size_t)16 * (2 * Hd + 8) * sizeof(_Float16);
    k_wmma_gemm_bt<<<dim3((Vd + 127) / 128, Bd / 16), 256, shOut, stream>>>(
        cat, w_out, b_out, lp, Vd, 2 * Hd);

    k_logsoftmax<<<Bd, 256, 0, stream>>>(lp);
}
